// OuterProductMean_42786464202817
// MI455X (gfx1250) — matrix operand order repaired and checked by reading, not compiled, not run on hardware
//
#include <hip/hip_runtime.h>

// ---------------------------------------------------------------------------
// OuterProductMean for MI455X (gfx1250), wave32 + WMMA(bf16) + TDM async loads
// Dims: N_SEQ=512, N_TOK=384, C_M=64, C=32, C_Z=128
// ---------------------------------------------------------------------------

typedef __bf16 v16bf __attribute__((ext_vector_type(16)));
typedef float  v8f   __attribute__((ext_vector_type(8)));
typedef unsigned int u32x4 __attribute__((ext_vector_type(4)));
typedef int          i32x8 __attribute__((ext_vector_type(8)));
typedef int          i32x4 __attribute__((ext_vector_type(4)));
typedef unsigned short u16;

#define N_SEQ 512
#define N_TOK 384
#define C_M   64
#define C_    32
#define C_Z   128
#define MBIG  (N_TOK * C_)      /* 12288 */
#define KBIG  N_SEQ             /* 512   */

static __device__ __forceinline__ u16 f2bf(float x) {
    union { float f; unsigned u; } v; v.f = x;
    unsigned r = (v.u + 0x7FFFu + ((v.u >> 16) & 1u)) >> 16;   // RNE
    return (u16)r;
}

union V16 { v16bf v; uint4 q[2]; };

// ---------------------------------------------------------------------------
// TDM: load a 2D bf16 tile (tile_rows x tile_k) from a row-major [rows][K]
// tensor in global memory into LDS (contiguous, row-major).  D# built per
// CDNA5 ISA ch.8 (group0 128b, group1 256b); groups 2/3 zero (2D tensor).
// 6-arg builtin form: (u32x4 g0, i32x8 g1, i32x4 g2, i32x4 g3, i32x8, cpol)
// ---------------------------------------------------------------------------
static __device__ __forceinline__ void tdm_load_tile(unsigned long long gaddr,
                                                     unsigned lds_off,
                                                     unsigned tile_k,
                                                     unsigned tile_rows,
                                                     unsigned tensor_k,
                                                     unsigned tensor_rows)
{
    u32x4 g0;
    g0[0] = 1u;                                            // count=1, user desc
    g0[1] = lds_off;                                       // lds_addr (bytes)
    g0[2] = (unsigned)(gaddr & 0xFFFFFFFFull);             // global_addr[31:0]
    g0[3] = (unsigned)((gaddr >> 32) & 0x01FFFFFFull)      // global_addr[56:32]
          | (2u << 30);                                    // type = 2 ("image")
    i32x8 g1;
    g1[0] = (int)(1u << 16);                               // data_size=1 (2B)
    g1[1] = (int)(tensor_k << 16);                         // tensor_dim0[15:0]
    g1[2] = (int)((tensor_k >> 16) | (tensor_rows << 16)); // dim0 hi | dim1 lo
    g1[3] = (int)((tensor_rows >> 16) | (tile_k << 16));   // dim1 hi | tile_dim0
    g1[4] = (int)tile_rows;                                // tile_dim1 (dim2=0)
    g1[5] = (int)tensor_k;                                 // dim0_stride[31:0]
    g1[6] = 0;                                             // dim0_stride hi
    g1[7] = 0;
    i32x4 z4; z4[0] = 0; z4[1] = 0; z4[2] = 0; z4[3] = 0;
    i32x8 z8;
    #pragma unroll
    for (int e = 0; e < 8; ++e) z8[e] = 0;
    __builtin_amdgcn_tensor_load_to_lds(g0, g1, z4, z4, z8, 0);
}

// ---------------------------------------------------------------------------
// Kernel 1: LayerNorm(C_M=64) + a = (m@Wa)*mask, b = (m@Wb)*mask.
// Output layout (bf16): aT[(i*32 + c)*512 + s], bT[(j*32 + d)*512 + s]
// (M-major, K=s contiguous -> WMMA A/B fragments are contiguous per lane).
// One wave handles (i, 16 consecutive s); each lane owns one output channel.
// ---------------------------------------------------------------------------
__global__ __launch_bounds__(256) void opm_ln_proj(
    const float* __restrict__ feat, const float* __restrict__ mask,
    const float* __restrict__ gamma, const float* __restrict__ beta,
    const float* __restrict__ Wa, const float* __restrict__ Wb,
    u16* __restrict__ aT, u16* __restrict__ bT)
{
    __shared__ float sWa[C_M * C_], sWb[C_M * C_], sG[C_M], sB[C_M];
    for (int e = threadIdx.x; e < C_M * C_; e += 256) { sWa[e] = Wa[e]; sWb[e] = Wb[e]; }
    if (threadIdx.x < C_M) { sG[threadIdx.x] = gamma[threadIdx.x]; sB[threadIdx.x] = beta[threadIdx.x]; }
    __syncthreads();

    const int w    = threadIdx.x >> 5;
    const int lane = threadIdx.x & 31;
    const int task = blockIdx.x * 8 + w;        // 384 * 32 = 12288 tasks
    const int i    = task >> 5;
    const int s0   = (task & 31) << 4;

    union { u16 u[16]; uint4 q[2]; } av, bv;

    for (int t = 0; t < 16; ++t) {
        const int s = s0 + t;
        const float2 x = ((const float2*)(feat + ((size_t)s * N_TOK + i) * C_M))[lane];
        float sum = x.x + x.y, sq = x.x * x.x + x.y * x.y;
        #pragma unroll
        for (int o = 16; o; o >>= 1) {
            sum += __shfl_xor(sum, o, 32);
            sq  += __shfl_xor(sq,  o, 32);
        }
        const float mu  = sum * (1.0f / 64.0f);
        const float var = sq * (1.0f / 64.0f) - mu * mu;
        const float rs  = rsqrtf(var + 1e-5f);
        const float m0  = (x.x - mu) * rs * sG[2 * lane]     + sB[2 * lane];
        const float m1  = (x.y - mu) * rs * sG[2 * lane + 1] + sB[2 * lane + 1];

        float a = 0.f, b = 0.f;
        #pragma unroll 8
        for (int k = 0; k < 32; ++k) {
            const float mk0 = __shfl(m0, k, 32);
            const float mk1 = __shfl(m1, k, 32);
            a += mk0 * sWa[(2 * k) * C_ + lane] + mk1 * sWa[(2 * k + 1) * C_ + lane];
            b += mk0 * sWb[(2 * k) * C_ + lane] + mk1 * sWb[(2 * k + 1) * C_ + lane];
        }
        const float mk = mask[(size_t)s * N_TOK + i];
        av.u[t] = f2bf(a * mk);
        bv.u[t] = f2bf(b * mk);
    }
    const size_t off = ((size_t)(i * C_ + lane)) * KBIG + s0;   // 32B-aligned
    *(uint4*)(aT + off)     = av.q[0];
    *(uint4*)(aT + off + 8) = av.q[1];
    *(uint4*)(bT + off)     = bv.q[0];
    *(uint4*)(bT + off + 8) = bv.q[1];
}

// ---------------------------------------------------------------------------
// Kernel 2: norm[i,j] = sum_s mask[s,i]*mask[s,j]  (384x384, mask fits in L2)
// ---------------------------------------------------------------------------
__global__ __launch_bounds__(256) void opm_norm(const float* __restrict__ mask,
                                                float* __restrict__ norm)
{
    const int idx = blockIdx.x * 256 + threadIdx.x;
    if (idx >= N_TOK * N_TOK) return;
    const int i = idx / N_TOK, j = idx % N_TOK;
    float acc = 0.f;
    for (int s = 0; s < N_SEQ; ++s)
        acc += mask[s * N_TOK + i] * mask[s * N_TOK + j];
    norm[idx] = acc;
}

// ---------------------------------------------------------------------------
// Kernel 3: fused big GEMM + Wout GEMM + epilogue.
//   Phase 1: C[128x128] = A[128x512] * B^T[128x512]^T  (bf16 WMMA, f32 acc)
//            A/B tiles streamed global->LDS by TDM (double-buffered, K-chunk 64)
//   Phase 2: reinterpret C as 16 (i,j)-pairs x 1024 (c*32+d); z = P @ Wout_bf16
//   Epilogue: z = (z + bout) / (norm + 1e-3)
// Block = 256 threads = 8 waves; grid = (96, 96).
// ---------------------------------------------------------------------------
__global__ __launch_bounds__(256) void opm_gemm_fused(
    const u16* __restrict__ aT, const u16* __restrict__ bT,
    const float* __restrict__ Wout, const float* __restrict__ bout,
    const float* __restrict__ norm, float* __restrict__ out)
{
    __shared__ __align__(16) unsigned char smem[65536];
    // phase 1: A dbl-buf @ [0, 32768), B dbl-buf @ [32768, 65536)
    // phase 2: P (16x1024 bf16) @ [0, 32768), Wbuf (128x128 bf16) @ [32768, ...)

    const int tid  = threadIdx.x;
    const int lane = tid & 31;
    const int w    = tid >> 5;
    const int wr   = w >> 1;          // wave tile-row group: rows wr*32 .. +32
    const int wc   = w & 1;           // wave tile-col group: cols wc*64 .. +64
    const int bi   = blockIdx.x;      // 4 i's per block
    const int bj   = blockIdx.y;      // 4 j's per block

    v8f acc[2][4];
    #pragma unroll
    for (int tr = 0; tr < 2; ++tr)
        #pragma unroll
        for (int tc = 0; tc < 4; ++tc)
            #pragma unroll
            for (int e = 0; e < 8; ++e) acc[tr][tc][e] = 0.f;

    const unsigned long long aG =
        (unsigned long long)(const void*)aT + (unsigned long long)(bi * 128) * KBIG * 2ull;
    const unsigned long long bG =
        (unsigned long long)(const void*)bT + (unsigned long long)(bj * 128) * KBIG * 2ull;
    const unsigned ldsA = (unsigned)(unsigned long long)(const void*)&smem[0];

    // ---- preload K-chunk 0 via TDM (wave 0 only: TDM issue ignores EXEC) ----
    if (w == 0) {
        tdm_load_tile(aG, ldsA +     0u, 64, 128, KBIG, MBIG);
        tdm_load_tile(bG, ldsA + 32768u, 64, 128, KBIG, MBIG);
        __builtin_amdgcn_s_wait_tensorcnt(0);
    }
    __syncthreads();

    // ---- phase 1: K = 512 in 8 chunks of 64, double-buffered ----
    for (int kc = 0; kc < 8; ++kc) {
        const int cur = kc & 1;
        if (w == 0 && kc < 7) {
            const unsigned nb = (unsigned)(cur ^ 1);
            const unsigned long long kb = (unsigned long long)(kc + 1) * 64ull * 2ull;
            tdm_load_tile(aG + kb, ldsA +          nb * 16384u, 64, 128, KBIG, MBIG);
            tdm_load_tile(bG + kb, ldsA + 32768u + nb * 16384u, 64, 128, KBIG, MBIG);
        }
        const u16* A = (const u16*)(smem + cur * 16384);
        const u16* B = (const u16*)(smem + 32768 + cur * 16384);

        #pragma unroll
        for (int ko = 0; ko < 64; ko += 32) {
            // A fragments: 16x32 bf16, lane = M row, split K chunks (ISA layout)
            V16 t;
            v16bf af[2];
            const int arow = wr * 32 + (lane & 15);
            const int aoff = ko + ((lane >> 4) << 3);
            #pragma unroll
            for (int tr = 0; tr < 2; ++tr) {
                const uint4* p = (const uint4*)(A + (arow + tr * 16) * 64 + aoff);
                t.q[0] = p[0];          // K = aoff .. aoff+7
                t.q[1] = p[2];          // K = aoff+16 .. aoff+23
                af[tr] = t.v;
            }
            // B fragments: 32x16 bf16, lane = N col, split K chunks (ISA layout)
            v16bf bf_[4];
            const int bcol = wc * 64 + (lane & 15);
            const int boff = ko + ((lane >> 4) << 3);
            #pragma unroll
            for (int tc = 0; tc < 4; ++tc) {
                const uint4* p = (const uint4*)(B + (bcol + tc * 16) * 64 + boff);
                t.q[0] = p[0];
                t.q[1] = p[2];
                bf_[tc] = t.v;
            }
            #pragma unroll
            for (int tr = 0; tr < 2; ++tr)
                #pragma unroll
                for (int tc = 0; tc < 4; ++tc)
                    acc[tr][tc] = __builtin_amdgcn_wmma_f32_16x16x32_bf16(
                        false, af[tr], false, bf_[tc], (short)0, acc[tr][tc],
                        false, false);
        }
        if (w == 0) __builtin_amdgcn_s_wait_tensorcnt(0);
        __syncthreads();
    }

    // ---- spill C tile to LDS as P[16 pairs][1024 = c*32+d] in bf16 ----
    u16* P = (u16*)smem;
    #pragma unroll
    for (int tr = 0; tr < 2; ++tr)
        #pragma unroll
        for (int tc = 0; tc < 4; ++tc)
            #pragma unroll
            for (int e = 0; e < 8; ++e) {
                const int r = wr * 32 + tr * 16 + e + ((lane >> 4) << 3);
                const int c = wc * 64 + tc * 16 + (lane & 15);
                const int pair = (r >> 5) * 4 + (c >> 5);
                const int k2   = ((r & 31) << 5) | (c & 31);
                P[pair * 1024 + k2] = f2bf(acc[tr][tc][e]);
            }

    // ---- phase 2: z[16 x 128] = P[16 x 1024] @ Wout[1024 x 128] ----
    u16* Wb = (u16*)(smem + 32768);
    v8f zacc;
    #pragma unroll
    for (int e = 0; e < 8; ++e) zacc[e] = 0.f;

    for (int c2 = 0; c2 < 8; ++c2) {          // K chunks of 128
        __syncthreads();                      // (c2==0: publishes P too)
        for (int e = tid; e < 128 * 128; e += 256) {
            const int kl = e >> 7, n = e & 127;
            Wb[n * 128 + kl] = f2bf(Wout[(c2 * 128 + kl) * C_Z + n]); // [n][k]
        }
        __syncthreads();
        #pragma unroll
        for (int ks = 0; ks < 4; ++ks) {
            const int kb = c2 * 128 + ks * 32;
            V16 t;
            // A2 fragment: rows = 16 pairs (shared by all waves)
            const uint4* pa = (const uint4*)(P + (lane & 15) * 1024 + kb + ((lane >> 4) << 3));
            t.q[0] = pa[0];
            t.q[1] = pa[2];
            const v16bf a2 = t.v;
            // B2 fragment: this wave's 16 output channels
            const int n = w * 16 + (lane & 15);
            const uint4* pb = (const uint4*)(Wb + n * 128 + ks * 32 + ((lane >> 4) << 3));
            t.q[0] = pb[0];
            t.q[1] = pb[2];
            zacc = __builtin_amdgcn_wmma_f32_16x16x32_bf16(
                false, a2, false, t.v, (short)0, zacc, false, false);
        }
    }

    // ---- epilogue: (z + bout) / (norm + 1e-3), write z[i][j][cz] ----
    const int cz = w * 16 + (lane & 15);
    const float bo = bout[cz];
    #pragma unroll
    for (int e = 0; e < 8; ++e) {
        const int pair = e + ((lane >> 4) << 3);
        const int ig = bi * 4 + (pair >> 2);
        const int jg = bj * 4 + (pair & 3);
        const float nv = norm[ig * N_TOK + jg] + 0.001f;
        out[((size_t)ig * N_TOK + jg) * C_Z + cz] = (zacc[e] + bo) / nv;
    }
}

// ---------------------------------------------------------------------------
extern "C" void kernel_launch(void* const* d_in, const int* in_sizes, int n_in,
                              void* d_out, int out_size, void* d_ws, size_t ws_size,
                              hipStream_t stream)
{
    const float* feat  = (const float*)d_in[0];
    const float* mask  = (const float*)d_in[1];
    const float* gamma = (const float*)d_in[2];
    const float* beta  = (const float*)d_in[3];
    const float* Wa    = (const float*)d_in[4];
    const float* Wb    = (const float*)d_in[5];
    const float* Wout  = (const float*)d_in[6];
    const float* bo    = (const float*)d_in[7];
    float* out = (float*)d_out;

    // workspace: aT (12 MB bf16) | bT (12 MB bf16) | norm (576 KB f32) ~ 25.7 MB
    u16* aT = (u16*)d_ws;
    u16* bT = aT + (size_t)MBIG * KBIG;
    float* nrm = (float*)(bT + (size_t)MBIG * KBIG);

    opm_ln_proj<<<1536, 256, 0, stream>>>(feat, mask, gamma, beta, Wa, Wb, aT, bT);
    opm_norm<<<(N_TOK * N_TOK + 255) / 256, 256, 0, stream>>>(mask, nrm);
    opm_gemm_fused<<<dim3(96, 96), 256, 0, stream>>>(aT, bT, Wout, bo, nrm, out);
}
